// Rotate_8_CCDeformConv2D_V1_5385888989768
// MI455X (gfx1250) — compile-verified, matrix operand order checked by reading, not run on hardware
//
#include <hip/hip_runtime.h>

// ---------------------------------------------------------------------------
// Deformable "rotate-8" conv for MI455X (gfx1250, wave32, WMMA).
//
// Pipeline (all on `stream`):
//   k_prep_tw   : build 8-orientation transformed weights, pre-shuffled into
//                 the v_wmma_f32_16x16x32_f16 B-operand lane layout (f16).
//   k_transpose : x NCHW -> NHWC so bilinear corners read 32 contiguous ch.
//   k_offsets   : 3x3 convs for offsets (18ch) + sigmoid mask (9ch).
//   k_main      : per 16-pixel tile: bilinear gather (f32) * mask -> f16 LDS,
//                 then sector-selected GEMM (M=16 px, N=32 co, K=288) via
//                 v_wmma_f32_16x16x32_f16, relu, write.
//
// group_mask is one-hot per pixel (angular sector) => only 1 of 8 orientation
// weight sets matters per pixel: 8x fewer WMMA flops than the dense einsum.
// ---------------------------------------------------------------------------

typedef __attribute__((ext_vector_type(16))) _Float16 v16h;
typedef __attribute__((ext_vector_type(8)))  _Float16 v8h;
typedef __attribute__((ext_vector_type(8)))  float    v8f;

#define HH   512
#define WW   512
#define HWSZ (HH * WW)
#define CIN  32
#define COUT 32
#define NORI 8
#define TAPS 9
#define CK   288          // 9 taps * 32 channels, ordered ck = tap*32 + ci
#define SROW 296          // padded LDS row (halves) to dodge bank conflicts

// ---------------- workspace layout (bytes) ----------------
// twB : 8*2*9*32*16 f16 = 147456
// offy/offx/modm : 9*HW f32 each
// xT  : HW*32 f32 (NHWC)
static constexpr size_t OFF_TWB  = 0;
static constexpr size_t OFF_OFFY = 147456;
static constexpr size_t OFF_OFFX = OFF_OFFY + (size_t)TAPS * HWSZ * 4;
static constexpr size_t OFF_MODM = OFF_OFFX + (size_t)TAPS * HWSZ * 4;
static constexpr size_t OFF_XT   = OFF_MODM + (size_t)TAPS * HWSZ * 4;

// ---------------- weight transform helpers ----------------
__device__ __forceinline__ float rotIdx(const float* __restrict__ M9, int r,
                                        int i, int j) {
  // value at (i,j) of rot90^r(M), numpy CCW: rot90(M)[i][j] = M[j][2-i]
  int ii, jj;
  switch (r & 3) {
    case 0:  ii = i;     jj = j;     break;
    case 1:  ii = j;     jj = 2 - i; break;
    case 2:  ii = 2 - i; jj = 2 - j; break;
    default: ii = 2 - j; jj = i;     break;
  }
  return M9[ii * 3 + jj];
}

__device__ __forceinline__ float rot45(const float* __restrict__ M9, int i, int j) {
  const float cs = 0.70710678118654752440f;  // cos(45)=sin(45)
  float ys = cs * (float)(i - 1) + cs * (float)(j - 1) + 1.0f;
  float xs = -cs * (float)(i - 1) + cs * (float)(j - 1) + 1.0f;
  float fy0 = floorf(ys), fx0 = floorf(xs);
  float wy = ys - fy0, wx = xs - fx0;
  int y0 = (int)fy0, x0 = (int)fx0;
  float v = 0.0f;
  for (int dy = 0; dy < 2; ++dy)
    for (int dx = 0; dx < 2; ++dx) {
      int iy = y0 + dy, ix = x0 + dx;
      if (iy >= 0 && iy < 3 && ix >= 0 && ix < 3) {
        float wgt = (dy ? wy : 1.0f - wy) * (dx ? wx : 1.0f - wx);
        v += M9[iy * 3 + ix] * wgt;
      }
    }
  return v;
}

// twB linear layout: [s][t][ks][lane][h]  (f16), matching the 16-bit B-operand
// lane layout of v_wmma_f32_16x16x32_f16: col = lane&15,
// K = (h/8)*16 + (lane>=16 ? 8 : 0) + (h%8);  contraction ck = ks*32 + K,
// with ck = tap*32 + ci  (so K-step == tap, within-step K == channel).
__global__ void k_prep_tw(const float* __restrict__ weight,
                          _Float16* __restrict__ twB) {
  int idx = blockIdx.x * blockDim.x + threadIdx.x;
  if (idx >= 8 * 2 * 9 * 32 * 16) return;
  int h  = idx & 15;
  int L  = (idx >> 4) & 31;
  int v  = idx >> 9;            // < 144
  int ks = v % 9;
  int t  = (v / 9) & 1;
  int s  = v / 18;
  int co = t * 16 + (L & 15);
  int Kf = ((h >> 3) << 4) + (((L >> 4) & 1) << 3) + (h & 7);  // = ci
  int ci = Kf;
  int tap = ks;
  int ty = tap / 3, tx = tap % 3;
  const float* M9 = weight + ((size_t)co * CIN + ci) * 9;  // (COUT,CIN,1,3,3)
  float val;
  if (s < 4) {
    val = rotIdx(M9, s, ty, tx);
  } else {
    int r = s - 4;  // rot90^r(w45) at (ty,tx) == w45 at rot-mapped index
    int ii, jj;
    switch (r) {
      case 0:  ii = ty;     jj = tx;     break;
      case 1:  ii = tx;     jj = 2 - ty; break;
      case 2:  ii = 2 - ty; jj = 2 - tx; break;
      default: ii = 2 - tx; jj = ty;     break;
    }
    val = rot45(M9, ii, jj);
  }
  twB[idx] = (_Float16)val;
}

// ---------------- NCHW -> NHWC ----------------
__global__ void k_transpose(const float* __restrict__ x, float* __restrict__ xT) {
  int pix = blockIdx.x * blockDim.x + threadIdx.x;
  if (pix >= HWSZ) return;
  float v[CIN];
#pragma unroll
  for (int c = 0; c < CIN; ++c) v[c] = x[(size_t)c * HWSZ + pix];
  float4* dst = (float4*)(xT + (size_t)pix * CIN);
#pragma unroll
  for (int q = 0; q < 8; ++q)
    dst[q] = make_float4(v[4 * q], v[4 * q + 1], v[4 * q + 2], v[4 * q + 3]);
}

// ---------------- offset / mask 3x3 convs ----------------
__global__ void k_offsets(const float* __restrict__ xT,
                          const float* __restrict__ offw,
                          const float* __restrict__ offb,
                          const float* __restrict__ maskw,
                          const float* __restrict__ maskb,
                          float* __restrict__ offy, float* __restrict__ offx,
                          float* __restrict__ modm) {
  __shared__ __align__(16) float wl[27 * CK];  // [oc][tap*32+ci]
  for (int i = threadIdx.x; i < 27 * CK; i += blockDim.x) {
    int oc = i / CK, rem = i % CK;
    int tap = rem >> 5, ci = rem & 31;
    wl[i] = (oc < 18) ? offw[((size_t)oc * CIN + ci) * 9 + tap]
                      : maskw[((size_t)(oc - 18) * CIN + ci) * 9 + tap];
  }
  __syncthreads();

  int pix = blockIdx.x * blockDim.x + threadIdx.x;
  if (pix >= HWSZ) return;
  int gy = pix >> 9, gx = pix & 511;

  float acc[27];
#pragma unroll
  for (int o = 0; o < 27; ++o) acc[o] = 0.0f;

  for (int tap = 0; tap < 9; ++tap) {
    int yy = gy + tap / 3 - 1, xx = gx + tap % 3 - 1;
    if ((unsigned)yy >= (unsigned)HH || (unsigned)xx >= (unsigned)WW) continue;
    const float4* src = (const float4*)(xT + ((size_t)yy * WW + xx) * CIN);
    float xv[CIN];
#pragma unroll
    for (int q = 0; q < 8; ++q) {
      float4 f = src[q];
      xv[4 * q] = f.x; xv[4 * q + 1] = f.y; xv[4 * q + 2] = f.z; xv[4 * q + 3] = f.w;
    }
#pragma unroll
    for (int o = 0; o < 27; ++o) {
      const float4* wrow = (const float4*)&wl[o * CK + tap * 32];
      float s = 0.0f;
#pragma unroll
      for (int q = 0; q < 8; ++q) {
        float4 w4 = wrow[q];
        s = fmaf(xv[4 * q], w4.x, s);
        s = fmaf(xv[4 * q + 1], w4.y, s);
        s = fmaf(xv[4 * q + 2], w4.z, s);
        s = fmaf(xv[4 * q + 3], w4.w, s);
      }
      acc[o] += s;
    }
  }
#pragma unroll
  for (int k = 0; k < 9; ++k) {
    float oy = (k == 4) ? 0.0f : acc[2 * k] + offb[2 * k];
    float ox = (k == 4) ? 0.0f : acc[2 * k + 1] + offb[2 * k + 1];
    float mv = (k == 4) ? 0.5f
                        : 1.0f / (1.0f + __expf(-(acc[18 + k] + maskb[k])));
    offy[(size_t)k * HWSZ + pix] = oy;
    offx[(size_t)k * HWSZ + pix] = ox;
    modm[(size_t)k * HWSZ + pix] = mv;
  }
}

// ---------------- main deformable conv (WMMA) ----------------
__global__ void __launch_bounds__(128)
k_main(const float* __restrict__ xT, const float* __restrict__ offy,
       const float* __restrict__ offx, const float* __restrict__ modm,
       const _Float16* __restrict__ twB, const float* __restrict__ gmask,
       float* __restrict__ out) {
  __shared__ __align__(16) _Float16 samp[4][16][SROW];  // [wave][pixel][ck]
  __shared__ int secArr[4][16];

  int lane = threadIdx.x & 31;
  int wv   = threadIdx.x >> 5;
  int gy   = blockIdx.y;
  int gx0  = (blockIdx.x * 4 + wv) * 16;

  // ---- phase 1: bilinear gather of sampled[16px][9tap][32ch] -> f16 LDS ----
  for (int t = lane; t < 144; t += 32) {
    int p = t / 9, tap = t % 9;
    int gx = gx0 + p;
    size_t pix = (size_t)gy * WW + gx;
    float oy = offy[(size_t)tap * HWSZ + pix];
    float ox = offx[(size_t)tap * HWSZ + pix];
    float m  = modm[(size_t)tap * HWSZ + pix];
    float py = (float)(gy - 1 + tap / 3) + oy;
    float px = (float)(gx - 1 + tap % 3) + ox;
    float fy0 = floorf(py), fx0 = floorf(px);
    float wy = py - fy0, wx = px - fx0;
    int y0 = (int)fy0, x0 = (int)fx0;

    float acc[CIN];
#pragma unroll
    for (int c = 0; c < CIN; ++c) acc[c] = 0.0f;

#pragma unroll
    for (int dy = 0; dy < 2; ++dy) {
      int iy = y0 + dy;
      if ((unsigned)iy >= (unsigned)HH) continue;
      float wyv = dy ? wy : 1.0f - wy;
#pragma unroll
      for (int dx = 0; dx < 2; ++dx) {
        int ix = x0 + dx;
        if ((unsigned)ix >= (unsigned)WW) continue;
        float wgt = wyv * (dx ? wx : 1.0f - wx);
        const float4* src = (const float4*)(xT + ((size_t)iy * WW + ix) * CIN);
#pragma unroll
        for (int q = 0; q < 8; ++q) {
          float4 f = src[q];
          acc[4 * q]     = fmaf(f.x, wgt, acc[4 * q]);
          acc[4 * q + 1] = fmaf(f.y, wgt, acc[4 * q + 1]);
          acc[4 * q + 2] = fmaf(f.z, wgt, acc[4 * q + 2]);
          acc[4 * q + 3] = fmaf(f.w, wgt, acc[4 * q + 3]);
        }
      }
    }
    _Float16* dst = &samp[wv][p][tap * 32];
#pragma unroll
    for (int c = 0; c < CIN; ++c) dst[c] = (_Float16)(acc[c] * m);
  }

  // ---- sector of each pixel from one-hot group_mask ----
  if (lane < 16) {
    size_t pix = (size_t)gy * WW + (gx0 + lane);
    int s = 0;
#pragma unroll
    for (int n = 0; n < NORI; ++n)
      if (gmask[(size_t)n * HWSZ + pix] > 0.5f) s = n;
    secArr[wv][lane] = s;
  }
  __syncthreads();  // LDS visibility (waves are independent; barrier is uniform)

  // ---- phase 2: sector-selected GEMM, M=16 px, N=32 co, K=288 ----
  int mySec = secArr[wv][lane & 15];
  int pixBase = (lane >> 4) * 8;
  const _Float16* arow = &samp[wv][lane & 15][0];
  int koff = (lane >> 4) * 8;  // K sub-offset per lane half (A-operand layout)

  for (int s = 0; s < NORI; ++s) {
    if (!__any(mySec == s)) continue;  // wave-uniform: EXEC stays all-ones
    v8f c0 = {};
    v8f c1 = {};
#pragma unroll
    for (int ks = 0; ks < 9; ++ks) {
      v8h lo = *(const v8h*)(arow + ks * 32 + koff);
      v8h hi = *(const v8h*)(arow + ks * 32 + koff + 16);
      v16h a = __builtin_shufflevector(lo, hi, 0, 1, 2, 3, 4, 5, 6, 7, 8, 9,
                                       10, 11, 12, 13, 14, 15);
      v16h b0 = *(const v16h*)(twB + ((((size_t)s * 2 + 0) * 9 + ks) * 32 + lane) * 16);
      v16h b1 = *(const v16h*)(twB + ((((size_t)s * 2 + 1) * 9 + ks) * 32 + lane) * 16);
      c0 = __builtin_amdgcn_wmma_f32_16x16x32_f16(false, a, false, b0,
                                                  (short)0, c0, false, false);
      c1 = __builtin_amdgcn_wmma_f32_16x16x32_f16(false, a, false, b1,
                                                  (short)0, c1, false, false);
    }
    // D layout: lanes 0-15: N=lane, M=r; lanes 16-31: N=lane-16, M=8+r
    int coBase = lane & 15;
#pragma unroll
    for (int r = 0; r < 8; ++r) {
      int p = pixBase + r;
      if (secArr[wv][p] == s) {
        size_t pi = (size_t)gy * WW + (gx0 + p);
        out[(size_t)coBase * HWSZ + pi]        = fmaxf(c0[r], 0.0f);
        out[(size_t)(coBase + 16) * HWSZ + pi] = fmaxf(c1[r], 0.0f);
      }
    }
  }
}

// ---------------------------------------------------------------------------
extern "C" void kernel_launch(void* const* d_in, const int* in_sizes, int n_in,
                              void* d_out, int out_size, void* d_ws,
                              size_t ws_size, hipStream_t stream) {
  (void)in_sizes; (void)n_in; (void)out_size; (void)ws_size;
  const float* x      = (const float*)d_in[0];  // (1,32,512,512)
  const float* weight = (const float*)d_in[1];  // (32,32,1,3,3)
  const float* offw   = (const float*)d_in[2];  // (18,32,3,3)
  const float* offb   = (const float*)d_in[3];  // (18,)
  const float* maskw  = (const float*)d_in[4];  // (9,32,3,3)
  const float* maskb  = (const float*)d_in[5];  // (9,)
  const float* gmask  = (const float*)d_in[6];  // (8,512,512)
  float* out = (float*)d_out;                   // (1,32,512,512)

  char* ws = (char*)d_ws;
  _Float16* twB = (_Float16*)(ws + OFF_TWB);
  float* offy   = (float*)(ws + OFF_OFFY);
  float* offx   = (float*)(ws + OFF_OFFX);
  float* modm   = (float*)(ws + OFF_MODM);
  float* xT     = (float*)(ws + OFF_XT);

  k_prep_tw<<<288, 256, 0, stream>>>(weight, twB);
  k_transpose<<<HWSZ / 256, 256, 0, stream>>>(x, xT);
  k_offsets<<<HWSZ / 256, 256, 0, stream>>>(xT, offw, offb, maskw, maskb,
                                            offy, offx, modm);
  k_main<<<dim3(WW / 64, HH), 128, 0, stream>>>(xT, offy, offx, modm, twB,
                                                gmask, out);
}
